// YOLOLossV2_1675037246085
// MI455X (gfx1250) — compile-verified, matrix operand order checked by reading, not run on hardware
//
#include <hip/hip_runtime.h>
#include <hip/hip_bf16.h>

typedef __attribute__((ext_vector_type(16))) _Float16 v16h;
typedef __attribute__((ext_vector_type(8)))  float    v8f;

#define BATCH 16
#define NT    32
#define NREC  (BATCH * NT)   // 512 target slots per scale
#define NCLS  80

// anchors / 640
__constant__ float c_anch[3][3][2] = {
    {{10.f/640.f, 13.f/640.f}, {16.f/640.f, 30.f/640.f}, {33.f/640.f, 23.f/640.f}},
    {{30.f/640.f, 61.f/640.f}, {62.f/640.f, 45.f/640.f}, {59.f/640.f,119.f/640.f}},
    {{116.f/640.f, 90.f/640.f},{156.f/640.f,198.f/640.f},{373.f/640.f,326.f/640.f}}};

// ws layout (floats):
//   [s*6 + 0..5] : box, cls, pos, neg, negfix, n_obj   (s = 0..2)  -> 18 floats
//   [32 + (s*NREC + r)*6 .. +5] : flat(-1 if unused), cls_id, tx, ty, tw, th

// ---------- math helpers ----------
__device__ __forceinline__ float logsig(float x) {
    // log(sigmoid(x)), numerically stable
    return (x < 0.f) ? (x - log1pf(expf(x))) : (-log1pf(expf(-x)));
}
__device__ __forceinline__ float focal01(float x, float t) {
    float p   = 1.f / (1.f + expf(-x));
    float bce = -(t * logsig(x) + (1.f - t) * logsig(-x));
    float pt  = t * p + (1.f - t) * (1.f - p);
    float om  = 1.f - pt;
    return bce * om * om;
}

// ---------- exact wave32 reduction via V_WMMA_F32_16X16X32_F16 ----------
// Split v into f16 hi+lo (products with 1.0 are exact, accumulate in f32).
// A rows m live in lanes m and m+16 (16 K-slots each); B = ones =>
// D[m][n] = 16*(v_m + v_{m+16}). Sum lane's 8 D regs + shfl_xor(16), /16.
__device__ __forceinline__ float wave_reduce_wmma(float v) {
    _Float16 hi = (_Float16)v;
    _Float16 lo = (_Float16)(v - (float)hi);
    v16h ah, al, bones;
#pragma unroll
    for (int i = 0; i < 16; ++i) { ah[i] = hi; al[i] = lo; bones[i] = (_Float16)1.0f; }
    v8f c = {};
    c = __builtin_amdgcn_wmma_f32_16x16x32_f16(false, ah, false, bones, (short)0, c, false, false);
    c = __builtin_amdgcn_wmma_f32_16x16x32_f16(false, al, false, bones, (short)0, c, false, false);
    float s = c[0] + c[1] + c[2] + c[3] + c[4] + c[5] + c[6] + c[7];
    s += __shfl_xor(s, 16, 32);
    return s * 0.0625f;
}

// ---------- kernels ----------
__global__ void k_zero(float* ws) {
    if (threadIdx.x < 18) ws[threadIdx.x] = 0.f;
}

__global__ void k_targets(const float* __restrict__ tgt, float* __restrict__ ws) {
    const int s   = blockIdx.x;           // scale
    const int tid = threadIdx.x;          // 0..511
    const int b   = tid / NT, t = tid % NT;
    const int Hs  = (s == 0) ? 80 : ((s == 1) ? 40 : 20);
    const int Ws  = Hs;

    __shared__ int sflat[NREC];

    const float* r = tgt + (b * NT + t) * 5;
    float cls = r[0], xc = r[1], yc = r[2], w = r[3], h = r[4];
    bool valid = (w > 0.f) && (h > 0.f);

    int best = 0; float bi = -1.f, bw = c_anch[s][0][0], bh = c_anch[s][0][1];
#pragma unroll
    for (int k = 0; k < 3; ++k) {
        float aw = c_anch[s][k][0], ah = c_anch[s][k][1];
        float inter = fminf(w, aw) * fminf(h, ah);
        float iou   = inter / (w * h + aw * ah - inter + 1e-6f);
        if (iou > bi) { bi = iou; best = k; bw = aw; bh = ah; }
    }
    int gi = (int)(xc * (float)Ws); gi = min(max(gi, 0), Ws - 1);
    int gj = (int)(yc * (float)Hs); gj = min(max(gj, 0), Hs - 1);
    int flat = valid ? (((b * 3 + best) * Hs + gj) * Ws + gi) : -1;

    sflat[tid] = flat;
    __syncthreads();

    bool sel = (flat >= 0);
    if (sel) {
        for (int j = tid + 1; j < NREC; ++j)
            if (sflat[j] == flat) { sel = false; break; }   // later target wins
    }

    float* rec = ws + 32 + (s * NREC + tid) * 6;
    rec[0] = sel ? (float)flat : -1.f;
    rec[1] = cls;
    rec[2] = xc * (float)Ws - (float)gi;
    rec[3] = yc * (float)Hs - (float)gj;
    rec[4] = logf(w / bw + 1e-6f);
    rec[5] = logf(h / bh + 1e-6f);
}

// Stream only the 48 objectness planes (channel a*85+4) of one scale.
__global__ void k_neg(const float* __restrict__ pred, int HW, int n, float* __restrict__ slot) {
    int i = blockIdx.x * blockDim.x + threadIdx.x;
    float acc = 0.f;
    if (i < n) {
        int plane = i / HW, within = i - plane * HW;
        int b = plane / 3, a = plane - 3 * b;
        float x = pred[(size_t)(b * 255 + a * 85 + 4) * HW + within];
        acc = focal01(x, 0.f);   // softplus(x) * sigmoid(x)^2
    }
    float tot = wave_reduce_wmma(acc);
    if ((threadIdx.x & 31) == 0) atomicAdd(slot, tot);
}

// One wave per target record: lanes cover the 85 channels of the positive cell.
__global__ void k_pos(const float* __restrict__ pred, int Hs, float* __restrict__ ws, int s) {
    const int wave = threadIdx.x >> 5, lane = threadIdx.x & 31;
    const int rIdx = blockIdx.x * 8 + wave;          // 0..511
    const int Ws = Hs, HW = Hs * Ws;

    const float* rec = ws + 32 + (s * NREC + rIdx) * 6;
    int   flat  = (int)rec[0];
    bool  sel   = (flat >= 0);
    int   clsid = (int)rec[1];
    float tx = rec[2], ty = rec[3], tw = rec[4], th = rec[5];

    int gi = 0, gj = 0, a = 0, b = 0;
    if (sel) {
        gi = flat % Ws; int t1 = flat / Ws;
        gj = t1 % Hs;   int t2 = t1 / Hs;
        a  = t2 % 3;    b  = t2 / 3;
    }

    float accBox = 0.f, accCls = 0.f, accPos = 0.f, accFix = 0.f;
    for (int c = lane; c < 85; c += 32) {
        if (sel) {
            float x = pred[(size_t)(b * 255 + a * 85 + c) * HW + gj * Ws + gi];
            if (c == 0) { float p = 1.f / (1.f + expf(-x)); float d = p - tx; accBox += d * d; }
            else if (c == 1) { float p = 1.f / (1.f + expf(-x)); float d = p - ty; accBox += d * d; }
            else if (c == 2) { float d = x - tw; accBox += d * d; }
            else if (c == 3) { float d = x - th; accBox += d * d; }
            else if (c == 4) { accPos += focal01(x, 1.f); accFix += focal01(x, 0.f); }
            else { float t = ((c - 5) == clsid) ? 1.f : 0.f; accCls += focal01(x, t); }
        }
    }

    float sBox = wave_reduce_wmma(accBox);
    float sCls = wave_reduce_wmma(accCls);
    float sPos = wave_reduce_wmma(accPos);
    float sFix = wave_reduce_wmma(accFix);

    if (lane == 0) {
        float* slots = ws + s * 6;
        atomicAdd(&slots[0], sBox);
        atomicAdd(&slots[1], sCls);
        atomicAdd(&slots[2], sPos);
        atomicAdd(&slots[4], sFix);
        if (sel) atomicAdd(&slots[5], 1.f);
    }
}

__global__ void k_final(const float* __restrict__ ws, float* __restrict__ out) {
    float total = 0.f;
    const int HWs[3] = {6400, 1600, 400};
#pragma unroll
    for (int s = 0; s < 3; ++s) {
        const float* sl = ws + s * 6;
        float box = sl[0], cls = sl[1], pos = sl[2], neg = sl[3], fix = sl[4], n = sl[5];
        float Ncells = 16.f * 3.f * (float)HWs[s];
        float nno    = Ncells - n;
        float has    = (n > 0.f) ? 1.f : 0.f;
        float l = 5.0f * box / fmaxf(2.f * n, 1.f) * has
                + 1.0f * pos / fmaxf(n, 1.f) * has
                + 0.5f * (neg - fix) / fmaxf(nno, 1.f)
                + 1.0f * cls / fmaxf(n * 80.f, 1.f) * has;
        total += l;
    }
    out[0] = total / 3.f;
}

extern "C" void kernel_launch(void* const* d_in, const int* in_sizes, int n_in,
                              void* d_out, int out_size, void* d_ws, size_t ws_size,
                              hipStream_t stream) {
    const float* p[3] = {(const float*)d_in[0], (const float*)d_in[1], (const float*)d_in[2]};
    const float* tg   = (const float*)d_in[3];
    float* ws  = (float*)d_ws;
    float* out = (float*)d_out;

    k_zero<<<1, 32, 0, stream>>>(ws);
    k_targets<<<3, 512, 0, stream>>>(tg, ws);

    const int Hs[3] = {80, 40, 20};
    for (int s = 0; s < 3; ++s) {
        int HW = Hs[s] * Hs[s];
        int n  = 48 * HW;                       // 16 batches * 3 anchors obj planes
        k_neg<<<(n + 255) / 256, 256, 0, stream>>>(p[s], HW, n, ws + s * 6 + 3);
        k_pos<<<64, 256, 0, stream>>>(p[s], Hs[s], ws, s);
    }
    k_final<<<1, 1, 0, stream>>>(ws, out);
}
